// SpatialWindowSelfAttention_50259707298464
// MI455X (gfx1250) — matrix operand order repaired and checked by reading, not compiled, not run on hardware
//
#include <hip/hip_runtime.h>
#include <hip/hip_bf16.h>

// CDNA5 wave32 WMMA types
typedef __bf16 v16bf __attribute__((ext_vector_type(16)));
typedef __bf16 v8bf  __attribute__((ext_vector_type(8)));
typedef __bf16 v4bf  __attribute__((ext_vector_type(4)));
typedef float  v8f   __attribute__((ext_vector_type(8)));

#define NHEAD 6

__device__ __forceinline__ v8f wmma_bf16(v16bf a, v16bf b, v8f c) {
  // D(f32 16x16) = A(16x32 bf16) * B(32x16 bf16) + C
  return __builtin_amdgcn_wmma_f32_16x16x32_bf16(false, a, false, b, (short)0, c,
                                                 false, false);
}

// Async global -> LDS copy (CDNA5, tracked by ASYNCcnt). LDS byte offset is the
// low 32 bits of the generic pointer (ISA: LDS aperture addr[31:0]).
__device__ __forceinline__ void async_ldg_b128(void* lds_dst, const void* gsrc) {
  unsigned ldsoff = (unsigned)(uintptr_t)lds_dst;
  asm volatile("global_load_async_to_lds_b128 %0, %1, off"
               :
               : "v"(ldsoff), "v"(gsrc)
               : "memory");
}
__device__ __forceinline__ void wait_async() {
  asm volatile("s_wait_asynccnt 0x0" ::: "memory");
}

// A-matrix 16x32 bf16 fragment: lane&15 = row M, lanes 0-15 hold K=0..15,
// lanes 16-31 hold K=16..31 (element s -> K = base+off+s).
__device__ __forceinline__ v16bf ld_a_frag(const __bf16* base, int stride, int m0,
                                           int kbase, int lane) {
  const __bf16* p = base + (size_t)(m0 + (lane & 15)) * stride + kbase +
                    ((lane >= 16) ? 16 : 0);
  v8bf lo = *(const v8bf*)p;
  v8bf hi = *(const v8bf*)(p + 8);
  v16bf r;
#pragma unroll
  for (int i = 0; i < 8; ++i) { r[i] = lo[i]; r[i + 8] = hi[i]; }
  return r;
}

// B-matrix 32x16 bf16 fragment from a K-contiguous (transposed, NxK) buffer:
// lane&15 = column N, lanes 0-15 hold K=0..15, lanes 16-31 hold K=16..31.
__device__ __forceinline__ v16bf ld_b_frag_t(const __bf16* base, int stride, int n0,
                                             int lane) {
  const __bf16* p = base + (size_t)(n0 + (lane & 15)) * stride +
                    ((lane >= 16) ? 16 : 0);
  v8bf lo = *(const v8bf*)p;
  v8bf hi = *(const v8bf*)(p + 8);
  v16bf r;
#pragma unroll
  for (int i = 0; i < 8; ++i) { r[i] = lo[i]; r[i + 8] = hi[i]; }
  return r;
}

// ---- Prep: convert + transpose weights to bf16 (done once per launch) ------
// wqkvT[n][k] (576x192), wpT[n][k] (192x192): k-contiguous so B frags are
// contiguous 16-element chunks.
__global__ __launch_bounds__(256) void prep_weights(const float* __restrict__ wqkv_w,
                                                    const float* __restrict__ wp_w,
                                                    __bf16* __restrict__ wqkvT,
                                                    __bf16* __restrict__ wpT) {
  int i = blockIdx.x * 256 + threadIdx.x;
  if (i < 576 * 192) {
    int n = i / 192, k = i - n * 192;
    wqkvT[i] = (__bf16)wqkv_w[(size_t)k * 576 + n];
  }
  int j = i - 576 * 192;
  if (j >= 0 && j < 192 * 192) {
    int n = j / 192, k = j - n * 192;
    wpT[j] = (__bf16)wp_w[(size_t)k * 192 + n];
  }
}

// One workgroup (8 wave32) per 8x8 window. Fully fused: QKV GEMM -> attention
// (bias+softmax) -> P@V -> output projection. LDS: 73728 (qkv bf16 64x576) +
// 73728 (union: x+Wt slab | score rows | Wp slab) = 147456 B (<160K, 2 WG/WGP).
__global__ __launch_bounds__(256) void swin_fused(
    const float* __restrict__ x, const __bf16* __restrict__ wqkvT,
    const float* __restrict__ wqkv_b, const __bf16* __restrict__ wpT,
    const float* __restrict__ wp_b, const float* __restrict__ bias_table,
    float* __restrict__ out) {
  __shared__ __attribute__((aligned(32))) char smem[73728 + 73728];
  __bf16* sQKV = (__bf16*)smem;           // [64][576] bf16: Q|K|V per token
  char*   uni  = smem + 73728;            // union region
  __bf16* sX   = (__bf16*)uni;            // [64][200] bf16 (phase 1)
  __bf16* sWt  = (__bf16*)(uni + 25600);  // [576][40] bf16 Wqkv slab (k-contig)

  const int tid   = threadIdx.x;
  const int wv    = tid >> 5;
  const int lane  = tid & 31;
  const int lhalf = (lane >= 16) ? 1 : 0;
  const int l16   = lane & 15;

  const int    wid     = blockIdx.x;       // 4608 windows
  const int    bimg    = wid / 576;
  const int    rem     = wid - bimg * 576;
  const int    wy      = rem / 24;
  const int    wx      = rem - wy * 24;
  const size_t rowbase = (size_t)bimg * 36864 + (size_t)wy * 8 * 192 + (size_t)wx * 8;

  // ---- Phase 0: gather x window -> LDS bf16 --------------------------------
  for (int e = tid * 4; e < 64 * 192; e += 256 * 4) {
    int t = e / 192;
    int c = e - t * 192;
    int iy = t >> 3, ix = t & 7;
    const float4 v = *(const float4*)(x + (rowbase + (size_t)iy * 192 + ix) * 192 + c);
    __bf16* d = sX + t * 200 + c;
    d[0] = (__bf16)v.x; d[1] = (__bf16)v.y; d[2] = (__bf16)v.z; d[3] = (__bf16)v.w;
  }

  // ---- Phase 1: QKV = X(64x192) @ Wqkv(192x576) ----------------------------
  const int m0  = 16 * (wv & 3);   // wave M strip
  const int n0g = 288 * (wv >> 2); // wave N group (18 tiles of 16)
  v8f accQ[18];
#pragma unroll
  for (int i = 0; i < 18; ++i) accQ[i] = (v8f){0, 0, 0, 0, 0, 0, 0, 0};

  for (int ks = 0; ks < 6; ++ks) {
    __syncthreads();
    // stage 576x32 bf16 weight slab via async global->LDS DMA (b128 chunks)
    {
      const __bf16* wsrc = wqkvT + (size_t)ks * 32;
      for (int c2 = tid; c2 < 576 * 4; c2 += 256) {
        int r = c2 >> 2, ch = c2 & 3;
        async_ldg_b128(sWt + r * 40 + ch * 8, wsrc + (size_t)r * 192 + ch * 8);
      }
      wait_async();
    }
    __syncthreads();
    v16bf a = ld_a_frag(sX, 200, m0, ks * 32, lane);
    // B fragments in groups of 3, all live before first WMMA of the group:
    // forces distinct registers -> waits become dscnt<=4 / <=2 / <=0, so LDS
    // loads overlap WMMA execution (DS returns are in-order per wave).
#pragma unroll
    for (int nt = 0; nt < 18; nt += 3) {
      v16bf bA = ld_b_frag_t(sWt, 40, n0g + (nt + 0) * 16, lane);
      v16bf bB = ld_b_frag_t(sWt, 40, n0g + (nt + 1) * 16, lane);
      v16bf bC = ld_b_frag_t(sWt, 40, n0g + (nt + 2) * 16, lane);
      accQ[nt + 0] = wmma_bf16(a, bA, accQ[nt + 0]);
      accQ[nt + 1] = wmma_bf16(a, bB, accQ[nt + 1]);
      accQ[nt + 2] = wmma_bf16(a, bC, accQ[nt + 2]);
    }
  }
  // epilogue: +bias, bf16, into sQKV  (C/D layout: VGPR i -> row m0+i (+8 hi half))
#pragma unroll
  for (int nt = 0; nt < 18; ++nt) {
    int   col = n0g + nt * 16 + l16;
    float bb  = wqkv_b[col];
#pragma unroll
    for (int i = 0; i < 8; ++i) {
      int r = m0 + i + lhalf * 8;
      sQKV[r * 576 + col] = (__bf16)(accQ[nt][i] + bb);
    }
  }
  __syncthreads();

  // ---- Phase 2: attention, one head per wave (waves 0..5) ------------------
  if (wv < NHEAD) {
    const int hh = wv;
    float* sS = (float*)(uni + hh * 12288);  // [32][96] f32 scores; P bf16 at +64 f32

    // B frags for Q@K^T: K stored token-major -> K^T is naturally N(=key)xK layout
    v16bf bK[4];
#pragma unroll
    for (int nt = 0; nt < 4; ++nt)
      bK[nt] = ld_b_frag_t(sQKV + 192 + hh * 32, 576, nt * 16, lane);

    // B frags for P@V: V[token k][dim n] -> strided gather per lane
    v16bf bV[4];
#pragma unroll
    for (int nt = 0; nt < 2; ++nt)
#pragma unroll
      for (int kk = 0; kk < 2; ++kk) {
        v16bf t;
        int ncol = 384 + hh * 32 + nt * 16 + l16;
        int kb   = kk * 32 + lhalf * 16;
#pragma unroll
        for (int e = 0; e < 16; ++e) t[e] = sQKV[(kb + e) * 576 + ncol];
        bV[nt * 2 + kk] = t;
      }

    const float scale = 0.17677669529663687f;  // 1/sqrt(32)
    for (int half = 0; half < 2; ++half) {
      const int mb = half * 32;
      // scores S = Q K^T * scale + rel-pos bias
#pragma unroll
      for (int mt = 0; mt < 2; ++mt) {
        v16bf a = ld_a_frag(sQKV, 576, mb + mt * 16, hh * 32, lane);
#pragma unroll
        for (int nt = 0; nt < 4; ++nt) {
          v8f acc = (v8f){0, 0, 0, 0, 0, 0, 0, 0};
          acc = wmma_bf16(a, bK[nt], acc);
          int kcol = nt * 16 + l16;
          int ky = kcol >> 3, kx = kcol & 7;
#pragma unroll
          for (int i = 0; i < 8; ++i) {
            int lr = mt * 16 + i + lhalf * 8;
            int q  = mb + lr;
            int qy = q >> 3, qx = q & 7;
            float bv = bias_table[((qy - ky + 7) * 15 + (qx - kx + 7)) * NHEAD + hh];
            sS[lr * 96 + kcol] = acc[i] * scale + bv;
          }
        }
      }
      // row softmax: one row per lane, vectorized float4; P bf16 beside f32 row
      {
        float* row = sS + lane * 96;
        float  mx  = -1e30f;
#pragma unroll
        for (int j = 0; j < 16; ++j) {
          float4 v = *(const float4*)(row + 4 * j);
          mx = fmaxf(mx, fmaxf(fmaxf(v.x, v.y), fmaxf(v.z, v.w)));
        }
        float sum = 0.f;
#pragma unroll
        for (int j = 0; j < 16; ++j) {
          float4 v = *(const float4*)(row + 4 * j);
          v.x = __expf(v.x - mx);
          v.y = __expf(v.y - mx);
          v.z = __expf(v.z - mx);
          v.w = __expf(v.w - mx);
          sum += (v.x + v.y) + (v.z + v.w);
          *(float4*)(row + 4 * j) = v;
        }
        float   inv  = 1.f / sum;
        __bf16* prow = (__bf16*)(row + 64);
#pragma unroll
        for (int j = 0; j < 16; ++j) {
          float4 v = *(const float4*)(row + 4 * j);
          v4bf  p;
          p[0] = (__bf16)(v.x * inv);
          p[1] = (__bf16)(v.y * inv);
          p[2] = (__bf16)(v.z * inv);
          p[3] = (__bf16)(v.w * inv);
          *(v4bf*)(prow + 4 * j) = p;  // ds_store_b64
        }
      }
      // Y = P(32x64) @ V(64x32); Y overwrites the dead Q region of sQKV
      const __bf16* sP = (const __bf16*)(uni + hh * 12288 + 256);
#pragma unroll
      for (int mt = 0; mt < 2; ++mt) {
        v16bf a0 = ld_a_frag(sP, 192, mt * 16, 0, lane);
        v16bf a1 = ld_a_frag(sP, 192, mt * 16, 32, lane);
#pragma unroll
        for (int nt = 0; nt < 2; ++nt) {
          v8f acc = (v8f){0, 0, 0, 0, 0, 0, 0, 0};
          acc = wmma_bf16(a0, bV[nt * 2 + 0], acc);
          acc = wmma_bf16(a1, bV[nt * 2 + 1], acc);
          int ncol = hh * 32 + nt * 16 + l16;
#pragma unroll
          for (int i = 0; i < 8; ++i) {
            int q = mb + mt * 16 + i + lhalf * 8;
            sQKV[q * 576 + ncol] = (__bf16)acc[i];
          }
        }
      }
    }
  }

  // ---- Phase 3: out = Y(64x192) @ Wp(192x192) + b --------------------------
  __bf16*   sWpT = (__bf16*)uni;  // [192][40] Wp slab (reuses score region)
  const int n0p  = 96 * (wv >> 2);
  v8f accO[6];
#pragma unroll
  for (int i = 0; i < 6; ++i) accO[i] = (v8f){0, 0, 0, 0, 0, 0, 0, 0};
  for (int ks = 0; ks < 6; ++ks) {
    __syncthreads();
    {
      const __bf16* wsrc = wpT + (size_t)ks * 32;
      for (int c2 = tid; c2 < 192 * 4; c2 += 256) {
        int r = c2 >> 2, ch = c2 & 3;
        async_ldg_b128(sWpT + r * 40 + ch * 8, wsrc + (size_t)r * 192 + ch * 8);
      }
      wait_async();
    }
    __syncthreads();
    v16bf a = ld_a_frag(sQKV, 576, m0, ks * 32, lane);
#pragma unroll
    for (int nt = 0; nt < 6; nt += 3) {
      v16bf bA = ld_b_frag_t(sWpT, 40, n0p + (nt + 0) * 16, lane);
      v16bf bB = ld_b_frag_t(sWpT, 40, n0p + (nt + 1) * 16, lane);
      v16bf bC = ld_b_frag_t(sWpT, 40, n0p + (nt + 2) * 16, lane);
      accO[nt + 0] = wmma_bf16(a, bA, accO[nt + 0]);
      accO[nt + 1] = wmma_bf16(a, bB, accO[nt + 1]);
      accO[nt + 2] = wmma_bf16(a, bC, accO[nt + 2]);
    }
  }
  // epilogue: +bias, scatter fp32 back to (b, h*w, c)
#pragma unroll
  for (int nt = 0; nt < 6; ++nt) {
    int   col = n0p + nt * 16 + l16;
    float bb  = wp_b[col];
#pragma unroll
    for (int i = 0; i < 8; ++i) {
      int t  = m0 + i + lhalf * 8;
      int iy = t >> 3, ix = t & 7;
      out[(rowbase + (size_t)iy * 192 + ix) * 192 + col] = accO[nt][i] + bb;
    }
  }
}

extern "C" void kernel_launch(void* const* d_in, const int* in_sizes, int n_in,
                              void* d_out, int out_size, void* d_ws, size_t ws_size,
                              hipStream_t stream) {
  (void)in_sizes; (void)n_in; (void)ws_size; (void)out_size;
  const float* x          = (const float*)d_in[0];
  const float* wqkv_w     = (const float*)d_in[1];
  const float* wqkv_b     = (const float*)d_in[2];
  const float* wp_w       = (const float*)d_in[3];
  const float* wp_b       = (const float*)d_in[4];
  const float* bias_table = (const float*)d_in[5];
  float*       out        = (float*)d_out;

  // workspace: transposed bf16 weights
  __bf16* wqkvT = (__bf16*)d_ws;                  // [576][192]
  __bf16* wpT   = wqkvT + 576 * 192;              // [192][192]

  prep_weights<<<dim3(576), dim3(256), 0, stream>>>(wqkv_w, wp_w, wqkvT, wpT);

  dim3 grid(4608), block(256);
  swin_fused<<<grid, block, 0, stream>>>(x, wqkvT, wqkv_b, wpT, wp_b, bias_table, out);
}
